// MAXLayerWithAttention_7962869366956
// MI455X (gfx1250) — compile-verified, matrix operand order checked by reading, not run on hardware
//
#include <hip/hip_runtime.h>
#include <hip/hip_bf16.h>
#include <math.h>

// Problem constants (B=8, S=2048, D=1024, U=1024, A=4)
#define MDIM 16384   // B*S
#define KDIM 1024    // D
#define NDIM 1024    // U
#define LDA  40      // padded LDS row stride in halves (32 data + 8 pad) -> conflict-free b128 reads

typedef __attribute__((ext_vector_type(16))) _Float16 v16h;
typedef __attribute__((ext_vector_type(8)))  _Float16 v8h;
typedef __attribute__((ext_vector_type(4)))  _Float16 v4h;
typedef __attribute__((ext_vector_type(8)))  float    v8f;

// ---------------------------------------------------------------------------
// Kernel 0: transpose + convert  w[D][U] f32  ->  wT[U][D] f16
// ---------------------------------------------------------------------------
__global__ __launch_bounds__(256) void k_transpose_w(const float* __restrict__ w,
                                                     _Float16* __restrict__ wT) {
    __shared__ float tile[32][33];
    const int bx = blockIdx.x & 31;        // n-tile (U/32)
    const int by = blockIdx.x >> 5;        // k-tile (D/32)
    const int tx = threadIdx.x & 31;
    const int ty = threadIdx.x >> 5;       // 0..7
#pragma unroll
    for (int j = 0; j < 4; ++j) {
        int k = by * 32 + ty + j * 8;
        tile[ty + j * 8][tx] = w[(size_t)k * NDIM + bx * 32 + tx];
    }
    __syncthreads();
#pragma unroll
    for (int j = 0; j < 4; ++j) {
        int n = bx * 32 + ty + j * 8;
        wT[(size_t)n * KDIM + by * 32 + tx] = (_Float16)tile[tx][ty + j * 8];
    }
}

// ---------------------------------------------------------------------------
// Kernel 1: attention  attn[m][a] = softmax_a( x[m,:] . w_att[:,a] + b_att[a] )
// One wave32 per row; coalesced x reads; butterfly shuffle reduction.
// ---------------------------------------------------------------------------
__global__ __launch_bounds__(256) void k_attention(const float* __restrict__ x,
                                                   const float* __restrict__ w_att,
                                                   const float* __restrict__ b_att,
                                                   float* __restrict__ attn) {
    const int wave = threadIdx.x >> 5;
    const int lane = threadIdx.x & 31;
    const int m = blockIdx.x * 8 + wave;
    const float* xr = x + (size_t)m * KDIM;

    float a0 = 0.f, a1 = 0.f, a2 = 0.f, a3 = 0.f;
    for (int k = lane; k < KDIM; k += 32) {
        float xv = xr[k];
        float4 wv = ((const float4*)w_att)[k];   // w_att row k (A=4 contiguous)
        a0 += xv * wv.x; a1 += xv * wv.y; a2 += xv * wv.z; a3 += xv * wv.w;
    }
#pragma unroll
    for (int off = 16; off > 0; off >>= 1) {
        a0 += __shfl_xor(a0, off, 32);
        a1 += __shfl_xor(a1, off, 32);
        a2 += __shfl_xor(a2, off, 32);
        a3 += __shfl_xor(a3, off, 32);
    }
    if (lane == 0) {
        a0 += b_att[0]; a1 += b_att[1]; a2 += b_att[2]; a3 += b_att[3];
        float mx = fmaxf(fmaxf(a0, a1), fmaxf(a2, a3));
        float e0 = __expf(a0 - mx), e1 = __expf(a1 - mx);
        float e2 = __expf(a2 - mx), e3 = __expf(a3 - mx);
        float inv = __builtin_amdgcn_rcpf(e0 + e1 + e2 + e3);
        ((float4*)attn)[m] = make_float4(e0 * inv, e1 * inv, e2 * inv, e3 * inv);
    }
}

// ---------------------------------------------------------------------------
// GEMM helpers
// ---------------------------------------------------------------------------
__device__ __forceinline__ void stage_tiles(const float* __restrict__ x,
                                            const _Float16* __restrict__ wT,
                                            _Float16* Ab, _Float16* Bb,
                                            int mBase, int nBase, int k0, int t) {
    // A tile: x[mBase..+127][k0..+31] f32 -> f16
    const int r0 = t >> 3;              // 0..31
    const int c4 = (t & 7) * 4;         // 0..28
#pragma unroll
    for (int p = 0; p < 4; ++p) {
        int r = p * 32 + r0;
        float4 v = *(const float4*)(x + (size_t)(mBase + r) * KDIM + k0 + c4);
        v4h hv;
        hv[0] = (_Float16)v.x; hv[1] = (_Float16)v.y;
        hv[2] = (_Float16)v.z; hv[3] = (_Float16)v.w;
        *(v4h*)&Ab[r * LDA + c4] = hv;
    }
    // B tile: wT[nBase..+127][k0..+31] f16
    const int n0 = t >> 2;              // 0..63
    const int c8 = (t & 3) * 8;         // 0..24
#pragma unroll
    for (int p = 0; p < 2; ++p) {
        int n = p * 64 + n0;
        *(v8h*)&Bb[n * LDA + c8] =
            *(const v8h*)(wT + (size_t)(nBase + n) * KDIM + k0 + c8);
    }
}

__device__ __forceinline__ void mma_step(const _Float16* Ab, const _Float16* Bb,
                                         v8f (&acc)[2][4],
                                         int wM, int wN, int h, int ln) {
    // ISA 16-bit A/B layout: lanes 0-15: K = 0..7 then 16..23 ;
    //                        lanes 16-31: K = 8..15 then 24..31
    v16h afrag[2], bfrag[4];
#pragma unroll
    for (int mi = 0; mi < 2; ++mi) {
        int m = wM * 32 + mi * 16 + ln;
        v8h lo = *(const v8h*)&Ab[m * LDA + 8 * h];
        v8h hi = *(const v8h*)&Ab[m * LDA + 16 + 8 * h];
        afrag[mi] = __builtin_shufflevector(lo, hi, 0,1,2,3,4,5,6,7,8,9,10,11,12,13,14,15);
    }
#pragma unroll
    for (int ni = 0; ni < 4; ++ni) {
        int n = wN * 64 + ni * 16 + ln;
        v8h lo = *(const v8h*)&Bb[n * LDA + 8 * h];
        v8h hi = *(const v8h*)&Bb[n * LDA + 16 + 8 * h];
        bfrag[ni] = __builtin_shufflevector(lo, hi, 0,1,2,3,4,5,6,7,8,9,10,11,12,13,14,15);
    }
#pragma unroll
    for (int mi = 0; mi < 2; ++mi)
#pragma unroll
        for (int ni = 0; ni < 4; ++ni)
            acc[mi][ni] = __builtin_amdgcn_wmma_f32_16x16x32_f16(
                false, afrag[mi], false, bfrag[ni],
                (short)0, acc[mi][ni], false, false);
}

// ---------------------------------------------------------------------------
// Kernel 2: fused  ws = x@w + b  (WMMA f16->f32, double-buffered LDS),
// then 4-way activation mix weighted by attn, store out[m][n] f32.
// 128x128 tile / block (256 thr = 8 waves, each wave 32x64 = 2x4 WMMA tiles).
// ---------------------------------------------------------------------------
__global__ __launch_bounds__(256) void k_gemm_act(const float*    __restrict__ x,
                                                  const _Float16* __restrict__ wT,
                                                  const float*    __restrict__ bias,
                                                  const float*    __restrict__ wf,
                                                  const float*    __restrict__ bf,
                                                  const float*    __restrict__ attn,
                                                  float*          __restrict__ out) {
    __shared__ _Float16 Alds[2][128 * LDA];   // [m][k] f16, padded rows
    __shared__ _Float16 Blds[2][128 * LDA];   // [n][k] f16, padded rows
    __shared__ float4   Atn[128];             // attention rows for this m-tile

    const int t     = threadIdx.x;
    const int nTile = blockIdx.x & 7;      // N / 128 = 8
    const int mTile = blockIdx.x >> 3;     // M / 128 = 128
    const int mBase = mTile * 128;
    const int nBase = nTile * 128;
    const int wave  = t >> 5;
    const int lane  = t & 31;
    const int wM    = wave & 3;            // 4 waves along M
    const int wN    = wave >> 2;           // 2 waves along N
    const int h     = lane >> 4;           // lane half (K sub-block select)
    const int ln    = lane & 15;           // row/col within 16

    if (t < 128) Atn[t] = ((const float4*)attn)[mBase + t];

    v8f acc[2][4] = {};

    // prologue: stage K-tile 0 into buffer 0
    stage_tiles(x, wT, Alds[0], Blds[0], mBase, nBase, 0, t);
    __syncthreads();

    // software-pipelined main loop: stage next buffer while consuming current
#pragma unroll 1
    for (int kt = 0; kt < KDIM / 32; kt += 2) {
        stage_tiles(x, wT, Alds[1], Blds[1], mBase, nBase, (kt + 1) * 32, t);
        mma_step(Alds[0], Blds[0], acc, wM, wN, h, ln);
        __syncthreads();
        if (kt + 2 < KDIM / 32)
            stage_tiles(x, wT, Alds[0], Blds[0], mBase, nBase, (kt + 2) * 32, t);
        mma_step(Alds[1], Blds[1], acc, wM, wN, h, ln);
        __syncthreads();
    }

    // ---- epilogue: bias + 4 activations weighted by attention ----
#pragma unroll
    for (int ni = 0; ni < 4; ++ni) {
        const int n_g = nBase + wN * 64 + ni * 16 + ln;
        const float  bn  = bias[n_g];
        const float4 wfv = ((const float4*)wf)[n_g];
        const float4 bfv = ((const float4*)bf)[n_g];
#pragma unroll
        for (int mi = 0; mi < 2; ++mi) {
#pragma unroll
            for (int i = 0; i < 8; ++i) {
                const int   mloc = wM * 32 + mi * 16 + 8 * h + i;
                const float wsv  = acc[mi][ni][i] + bn;
                const float4 at  = Atn[mloc];
                const float t0 = wsv * wfv.x + bfv.x;
                const float t1 = wsv * wfv.y + bfv.y;
                const float t2 = wsv * wfv.z + bfv.z;
                const float t3 = wsv * wfv.w + bfv.w;
                // relu
                float r = fmaxf(t0, 0.f) * at.x;
                // sigmoid: 1/(1+e^-t) via v_exp + v_rcp (straight-line, ~1ulp)
                r += at.y * __builtin_amdgcn_rcpf(1.f + __expf(-t1));
                // tanh: 1 - 2/(e^{2t}+1)  (saturates correctly at +/-inf)
                r += at.z * fmaf(-2.f, __builtin_amdgcn_rcpf(__expf(2.f * t2) + 1.f), 1.f);
                // exact gelu: 0.5*t*(1+erf(t/sqrt(2)))
                r += at.w * (0.5f * t3 * (1.f + erff(t3 * 0.7071067811865475f)));
                out[(size_t)(mBase + mloc) * NDIM + n_g] = r;
            }
        }
    }
}

// ---------------------------------------------------------------------------
extern "C" void kernel_launch(void* const* d_in, const int* in_sizes, int n_in,
                              void* d_out, int out_size, void* d_ws, size_t ws_size,
                              hipStream_t stream) {
    (void)in_sizes; (void)n_in; (void)out_size; (void)ws_size;
    const float* x     = (const float*)d_in[0];   // [B,S,D]
    const float* w     = (const float*)d_in[1];   // [D,U]
    const float* b     = (const float*)d_in[2];   // [U]
    const float* wf    = (const float*)d_in[3];   // [U,A]
    const float* bf    = (const float*)d_in[4];   // [U,A]
    const float* w_att = (const float*)d_in[5];   // [D,A]
    const float* b_att = (const float*)d_in[6];   // [A]
    float* out = (float*)d_out;                   // [B,S,U]

    // workspace layout: wT (U*K f16 = 2MB) | attn (M*4 f32 = 256KB)
    _Float16* wT   = (_Float16*)d_ws;
    float*    attn = (float*)((char*)d_ws + (size_t)NDIM * KDIM * sizeof(_Float16));

    k_transpose_w<<<(KDIM / 32) * (NDIM / 32), 256, 0, stream>>>(w, wT);
    k_attention<<<MDIM / 8, 256, 0, stream>>>(x, w_att, b_att, attn);
    k_gemm_act<<<(MDIM / 128) * (NDIM / 128), 256, 0, stream>>>(x, wT, b, wf, bf, attn, out);
}
